// GeneralizedRCNNMEGA_61701500175296
// MI455X (gfx1250) — matrix operand order repaired and checked by reading, not compiled, not run on hardware
//
#include <hip/hip_runtime.h>

// ---------------- CDNA5 WMMA types ----------------
typedef __bf16 v16bf __attribute__((ext_vector_type(16)));
typedef float  v8f   __attribute__((ext_vector_type(8)));

#define HW       3840      // 48*80 queries
#define DIMK     1024
#define NKEY     8224      // 11*384 + 2000 + 2000
#define NSEL_TOT 4224
#define NMEM     2000
#define NH       16
#define DH       64

union Frag {
  v16bf v;
  uint4 q[2];
  unsigned int w[8];
  unsigned short u[16];
};

__device__ __forceinline__ unsigned short f2bf(float f) {
  unsigned int u = __float_as_uint(f);
  u += 0x7FFFu + ((u >> 16) & 1u);   // round-to-nearest-even
  return (unsigned short)(u >> 16);
}

// One wave-level async copy: 32 lanes x 16B -> 512B global->LDS, tracked by ASYNCcnt.
__device__ __forceinline__ void async_ld16(unsigned lds_off, const unsigned short* gptr) {
  asm volatile("global_load_async_to_lds_b128 %0, %1, off"
               :: "v"(lds_off), "v"((unsigned long long)(uintptr_t)gptr)
               : "memory");
}
__device__ __forceinline__ void wait_async0() {
  asm volatile("s_wait_asynccnt 0" ::: "memory");
}

// ---------------- prep: q_in (f32 + bf16), transposed to [hw, dim] ----------------
__global__ void build_q(const float* __restrict__ fc, const float* __restrict__ ps,
                        float* __restrict__ q_in, unsigned short* __restrict__ q16) {
  int idx = blockIdx.x * blockDim.x + threadIdx.x;       // idx = d*HW + p (coalesced reads)
  if (idx >= DIMK * HW) return;
  int d = idx / HW, p = idx % HW;
  float v = fc[idx] + ps[idx];
  q_in[(size_t)p * DIMK + d] = v;
  q16[(size_t)p * DIMK + d]  = f2bf(v);
}

// ---------------- prep: k_in bf16 [NKEY, DIM] ----------------
__global__ void build_k(const float* __restrict__ fr, const float* __restrict__ ps,
                        const float* __restrict__ em, const float* __restrict__ gc,
                        const int* __restrict__ sel, const int* __restrict__ mi,
                        const int* __restrict__ gi, unsigned short* __restrict__ k16) {
  long idx = (long)blockIdx.x * blockDim.x + threadIdx.x;  // idx = r*DIM + d
  if (idx >= (long)NKEY * DIMK) return;
  int r = (int)(idx >> 10);
  int d = (int)(idx & 1023);
  float v;
  if (r < NSEL_TOT) {
    int f = r / 384;
    int p = sel[r];
    v = fr[((long)f * DIMK + d) * HW + p] + ps[(long)d * HW + p];
  } else if (r < NSEL_TOT + NMEM) {
    v = em[(long)mi[r - NSEL_TOT] * DIMK + d];
  } else {
    v = gc[(long)gi[r - NSEL_TOT - NMEM] * DIMK + d];
  }
  k16[idx] = f2bf(v);
}

// ---------------- prep: weight transpose to bf16 Wt[n][k] ----------------
__global__ void transpose_w(const float* __restrict__ W, unsigned short* __restrict__ Wt) {
  int idx = blockIdx.x * blockDim.x + threadIdx.x;         // idx = n*DIM + k
  if (idx >= DIMK * DIMK) return;
  int n = idx >> 10, k = idx & 1023;
  Wt[idx] = f2bf(W[(size_t)k * DIMK + n]);
}

// ---------------- WMMA GEMM: C[M,1024] = A[M,1024] @ Wt^T ----------------
// MODE 0: C16 row-major bf16       (Q, K projections)
// MODE 1: C16[n*ldT + m] bf16      (V^T)
// MODE 2: Cf[n*ldT + m] = acc + resid[m*1024+n]  (final out, transposed + residual)
template <int MODE>
__global__ __launch_bounds__(256) void gemm_bf16(
    const unsigned short* __restrict__ A, const unsigned short* __restrict__ Bt,
    unsigned short* __restrict__ C16, float* __restrict__ Cf,
    const float* __restrict__ resid, int M, int ldT) {
  int lane = threadIdx.x & 31;
  int wave = (int)((blockIdx.x * blockDim.x + threadIdx.x) >> 5);
  int l16 = lane & 15, hi = lane >> 4;
  int mt = wave >> 5;          // 1024/32 = 32 N-tiles
  int nt = wave & 31;
  int m0 = mt * 32, n0 = nt * 32;
  if (m0 >= M) return;

  v8f acc[2][2];
  for (int i = 0; i < 2; i++)
    for (int j = 0; j < 2; j++)
      for (int e = 0; e < 8; e++) acc[i][j][e] = 0.f;

  for (int kb = 0; kb < DIMK; kb += 32) {
    Frag a[2], b[2];
    for (int ms = 0; ms < 2; ms++) {
      // A 16x32 layout: lane<16 holds k {0..7,16..23}, lane>=16 {8..15,24..31}
      const unsigned short* ar = A + (size_t)(m0 + ms * 16 + l16) * DIMK + kb + hi * 8;
      a[ms].q[0] = *(const uint4*)ar;
      a[ms].q[1] = *(const uint4*)(ar + 16);
    }
    for (int ns = 0; ns < 2; ns++) {
      // B 32x16 layout: lane<16 holds k {0..7,16..23}, lane>=16 {8..15,24..31}
      const unsigned short* br = Bt + (size_t)(n0 + ns * 16 + l16) * DIMK + kb + hi * 8;
      b[ns].q[0] = *(const uint4*)br;
      b[ns].q[1] = *(const uint4*)(br + 16);
    }
    for (int ms = 0; ms < 2; ms++)
      for (int ns = 0; ns < 2; ns++)
        acc[ms][ns] = __builtin_amdgcn_wmma_f32_16x16x32_bf16(
            false, a[ms].v, false, b[ns].v, (short)0, acc[ms][ns], false, false);
  }

  for (int ms = 0; ms < 2; ms++)
    for (int ns = 0; ns < 2; ns++)
      for (int r = 0; r < 8; r++) {
        int m = m0 + ms * 16 + r + 8 * hi;   // C layout: VGPR r -> row r (+8 for hi lanes)
        int n = n0 + ns * 16 + l16;
        float val = acc[ms][ns][r];
        if (MODE == 0) C16[(size_t)m * DIMK + n] = f2bf(val);
        else if (MODE == 1) C16[(size_t)n * ldT + m] = f2bf(val);
        else Cf[(size_t)n * ldT + m] = val + resid[(size_t)m * DIMK + n];
      }
}

// ---------------- LDS-staged flash attention ----------------
// Block = 256 threads = 8 waves, all on ONE head, eight consecutive 16-query tiles.
// Per 32-key step the block double-buffers K (32x64 bf16) and V^T (64x32 bf16) in LDS
// via global_load_async_to_lds_b128 (ASYNCcnt), cutting K/V L2 traffic 8x.
// Per wave per step: S^T = K x Q^T (4 WMMAs), online softmax, ctx^T += V^T x P^T (4 WMMAs).
__global__ __launch_bounds__(256) void attn(
    const unsigned short* __restrict__ Q16, const unsigned short* __restrict__ K16,
    const unsigned short* __restrict__ Vt16, unsigned short* __restrict__ ctx16) {
  __shared__ __align__(16) unsigned short Kb[2][32][64];   // [buf][k][d]
  __shared__ __align__(16) unsigned short Vb[2][64][32];   // [buf][d][k]

  int tid  = threadIdx.x;
  int lane = tid & 31, w = tid >> 5;
  int l16 = lane & 15, hi = lane >> 4;
  int h  = blockIdx.x & 15;
  int qt = (blockIdx.x >> 4) * 8 + w;     // 0..239

  // B fragments of Q^T (loaded once): lane = column q, d {0..7,16..23} / {8..15,24..31} per half
  Frag bq[2];
  {
    const unsigned short* qr = Q16 + (size_t)(qt * 16 + l16) * DIMK + h * DH;
    for (int df = 0; df < 2; df++) {
      const unsigned short* p = qr + df * 32 + hi * 8;
      bq[df].q[0] = *(const uint4*)p;
      bq[df].q[1] = *(const uint4*)(p + 16);
    }
  }

  // async stage of one 32-key slice of K and V^T into LDS buffer b
  auto stage = [&](int b, int kb) {
    int i = tid >> 3, c = tid & 7;        // 32 rows x 8 chunks of 16B (K)
    async_ld16((unsigned)(uintptr_t)&Kb[b][i][c * 8],
               K16 + (size_t)(kb + i) * DIMK + h * DH + c * 8);
    int d2 = tid >> 2, c2 = tid & 3;      // 64 rows x 4 chunks of 16B (V^T)
    async_ld16((unsigned)(uintptr_t)&Vb[b][d2][c2 * 8],
               Vt16 + (size_t)(h * DH + d2) * NKEY + kb + c2 * 8);
  };

  v8f acc[4];
  for (int t = 0; t < 4; t++)
    for (int e = 0; e < 8; e++) acc[t][e] = 0.f;
  float m_run = -1e30f, l_run = 0.f;

  stage(0, 0);

  for (int kc = 0; kc < NKEY / 32; kc++) {
    int cur = kc & 1;
    wait_async0();           // this wave's async stage-writes for 'cur' have landed
    __syncthreads();         // every wave's writes landed -> buffer 'cur' complete
    if (kc + 1 < NKEY / 32) stage(cur ^ 1, (kc + 1) * 32);   // prefetch next slice

    // S^T tiles from LDS (ds_load_b128 fragments)
    v8f s[2];
    for (int t = 0; t < 2; t++) {
      for (int e = 0; e < 8; e++) s[t][e] = 0.f;
      for (int dh2 = 0; dh2 < 2; dh2++) {
        Frag ak;
        const unsigned short* p = &Kb[cur][t * 16 + l16][dh2 * 32 + hi * 8];
        ak.q[0] = *(const uint4*)p;
        ak.q[1] = *(const uint4*)(p + 16);
        s[t] = __builtin_amdgcn_wmma_f32_16x16x32_bf16(
            false, ak.v, false, bq[dh2].v, (short)0, s[t], false, false);
      }
    }

    // online softmax; lane holds 16 logits for query q = qt*16 + l16
    float mx = -1e30f;
    for (int t = 0; t < 2; t++)
      for (int e = 0; e < 8; e++) { s[t][e] *= 0.125f; mx = fmaxf(mx, s[t][e]); }
    mx = fmaxf(mx, __shfl_xor(mx, 16, 32));
    float m_new = fmaxf(m_run, mx);
    float alpha = __expf(m_run - m_new);
    float rs = 0.f;
    for (int t = 0; t < 2; t++)
      for (int e = 0; e < 8; e++) { float pv = __expf(s[t][e] - m_new); s[t][e] = pv; rs += pv; }
    rs += __shfl_xor(rs, 16, 32);
    l_run = l_run * alpha + rs;
    m_run = m_new;

    // build P^T as a B operand: C-layout of S^T matches after swapping the
    // k in [8,16)/[16,24) halves between lane L and L^16.
    unsigned int t0p[4], t1p[4];
    for (int j = 0; j < 4; j++) {
      t0p[j] = (unsigned)f2bf(s[0][2 * j]) | ((unsigned)f2bf(s[0][2 * j + 1]) << 16);
      t1p[j] = (unsigned)f2bf(s[1][2 * j]) | ((unsigned)f2bf(s[1][2 * j + 1]) << 16);
    }
    Frag pb;
    for (int j = 0; j < 4; j++) {
      unsigned int o0 = (unsigned int)__shfl_xor((int)t0p[j], 16, 32);
      unsigned int o1 = (unsigned int)__shfl_xor((int)t1p[j], 16, 32);
      pb.w[j]     = hi ? o1 : t0p[j];   // lane<16: k 0..7   ; lane>=16: k 16..23
      pb.w[4 + j] = hi ? t1p[j] : o0;   // lane<16: k 8..15  ; lane>=16: k 24..31
    }

    // rescale running ctx^T and accumulate V^T x P^T from LDS
    for (int dt = 0; dt < 4; dt++) {
      for (int e = 0; e < 8; e++) acc[dt][e] *= alpha;
      Frag av;
      const unsigned short* p = &Vb[cur][dt * 16 + l16][hi * 16];
      av.q[0] = *(const uint4*)p;
      av.q[1] = *(const uint4*)(p + 8);
      acc[dt] = __builtin_amdgcn_wmma_f32_16x16x32_bf16(
          false, av.v, false, pb.v, (short)0, acc[dt], false, false);
    }

    __syncthreads();         // all waves done reading 'cur' before it is restaged
  }

  float inv = 1.0f / l_run;
  int q = qt * 16 + l16;
  for (int dt = 0; dt < 4; dt++)
    for (int r = 0; r < 8; r++) {
      int d = dt * 16 + r + 8 * hi;
      ctx16[(size_t)q * DIMK + h * DH + d] = f2bf(acc[dt][r] * inv);
    }
}

// ---------------- host orchestration ----------------
extern "C" void kernel_launch(void* const* d_in, const int* in_sizes, int n_in,
                              void* d_out, int out_size, void* d_ws, size_t ws_size,
                              hipStream_t stream) {
  (void)in_sizes; (void)n_in; (void)out_size; (void)ws_size;
  const float* feats_cur  = (const float*)d_in[0];
  const float* ps         = (const float*)d_in[1];
  const float* feats_ref  = (const float*)d_in[2];
  const float* ext_mem    = (const float*)d_in[3];
  const float* glob_cache = (const float*)d_in[4];
  const float* Wq         = (const float*)d_in[5];
  const float* Wk         = (const float*)d_in[6];
  const float* Wv         = (const float*)d_in[7];
  const float* Wo         = (const float*)d_in[8];
  const int*   sel        = (const int*)d_in[9];
  const int*   mi         = (const int*)d_in[10];
  const int*   gi         = (const int*)d_in[11];
  float* out = (float*)d_out;

  char* w = (char*)d_ws;
  auto carve = [&](size_t bytes) {
    char* p = w;
    w += (bytes + 255) & ~(size_t)255;
    return p;
  };
  float*          q_in = (float*)carve((size_t)HW * DIMK * 4);
  unsigned short* q16  = (unsigned short*)carve((size_t)HW * DIMK * 2);
  unsigned short* k16  = (unsigned short*)carve((size_t)NKEY * DIMK * 2);
  unsigned short* Wqt  = (unsigned short*)carve((size_t)DIMK * DIMK * 2);
  unsigned short* Wkt  = (unsigned short*)carve((size_t)DIMK * DIMK * 2);
  unsigned short* Wvt  = (unsigned short*)carve((size_t)DIMK * DIMK * 2);
  unsigned short* Wot  = (unsigned short*)carve((size_t)DIMK * DIMK * 2);
  unsigned short* Qp   = (unsigned short*)carve((size_t)HW * DIMK * 2);
  unsigned short* Kp   = (unsigned short*)carve((size_t)NKEY * DIMK * 2);
  unsigned short* Vt   = (unsigned short*)carve((size_t)DIMK * NKEY * 2);
  unsigned short* ctx  = (unsigned short*)carve((size_t)HW * DIMK * 2);

  // prep
  build_q<<<(HW * DIMK + 255) / 256, 256, 0, stream>>>(feats_cur, ps, q_in, q16);
  build_k<<<(int)(((long)NKEY * DIMK + 255) / 256), 256, 0, stream>>>(
      feats_ref, ps, ext_mem, glob_cache, sel, mi, gi, k16);
  transpose_w<<<(DIMK * DIMK + 255) / 256, 256, 0, stream>>>(Wq, Wqt);
  transpose_w<<<(DIMK * DIMK + 255) / 256, 256, 0, stream>>>(Wk, Wkt);
  transpose_w<<<(DIMK * DIMK + 255) / 256, 256, 0, stream>>>(Wv, Wvt);
  transpose_w<<<(DIMK * DIMK + 255) / 256, 256, 0, stream>>>(Wo, Wot);

  // projections: waves = (M/32)*(1024/32); 8 waves / 256-thread block
  gemm_bf16<0><<<(HW / 32) * 32 / 8, 256, 0, stream>>>(q16, Wqt, Qp, nullptr, nullptr, HW, 0);
  gemm_bf16<0><<<(NKEY / 32) * 32 / 8, 256, 0, stream>>>(k16, Wkt, Kp, nullptr, nullptr, NKEY, 0);
  gemm_bf16<1><<<(NKEY / 32) * 32 / 8, 256, 0, stream>>>(k16, Wvt, Vt, nullptr, nullptr, NKEY, NKEY);

  // flash attention: 16 heads x 30 blocks, 8 q-tiles per block
  attn<<<NH * (HW / 16 / 8), 256, 0, stream>>>(Qp, Kp, Vt, ctx);

  // out = q_in + ctx @ Wo, stored transposed as [dim, hw]
  gemm_bf16<2><<<(HW / 32) * 32 / 8, 256, 0, stream>>>(ctx, Wot, nullptr, out, q_in, HW, HW);
}
